// TrustDiffuserModule_38362647888262
// MI455X (gfx1250) — compile-verified, matrix operand order checked by reading, not run on hardware
//
#include <hip/hip_runtime.h>

#define NB_REFC 10
#define BATCH   8
#define NTOK    197
#define DIM     768
#define HEADS   12
#define HD      64
#define STEPSC  3

// padded attention dims / LDS strides (elements)
#define NP   224   // 197 padded to 14*16 (and 7*32 for K loops)
#define QSS  72    // q/k row stride (144 B = 128 B data + 16 B TDM pad)
#define VST  232   // v^T row stride
#define AST  232   // A / S row stride

typedef __bf16 bf16;
typedef __attribute__((ext_vector_type(16))) __bf16 v16bf;
typedef __attribute__((ext_vector_type(8)))  __bf16 v8bf;
typedef __attribute__((ext_vector_type(8)))  float  v8f;
typedef __attribute__((ext_vector_type(4)))  unsigned int v4u;
typedef __attribute__((ext_vector_type(8)))  int v8i;
typedef __attribute__((ext_vector_type(4)))  int v4i;

union FragBF { v16bf v; v8bf h[2]; };
union Pack8  { v8bf v; bf16 e[8]; };

__device__ __forceinline__ bf16 f2bf(float f) {
  union { float f; unsigned u; } c; c.f = f;
  unsigned u = c.u + 0x7FFFu + ((c.u >> 16) & 1u);   // round-to-nearest-even
  union { unsigned short s; bf16 b; } o; o.s = (unsigned short)(u >> 16);
  return o.b;
}

// 16x32 bf16 WMMA fragment from row-major LDS (ISA 7.12.2): two 16B loads/lane.
__device__ __forceinline__ v16bf load_frag(const bf16* base, int stride, int rowbase, int kbase) {
  const int lane = threadIdx.x & 31;
  const bf16* p = base + (rowbase + (lane & 15)) * stride + kbase + ((lane >> 4) << 3);
  FragBF f;
  f.h[0] = *(const v8bf*)(p);
  f.h[1] = *(const v8bf*)(p + 16);
  return f.v;
}

// TDM 2D bf16 tile load: rows_valid real rows (OOB rows read as zero up to
// tile_rows), row_elems elements/row, global row stride in elements.
// LDS pad: pad_int_code -> pad after 2<<code DWORDs; pad_amt_code -> (code+1) DWORDs.
__device__ __forceinline__ void tdm_load_2d_bf16(
    unsigned lds_addr, const bf16* gbase, unsigned rows_valid, unsigned row_elems,
    unsigned row_stride, unsigned tile_rows, unsigned pad_int_code, unsigned pad_amt_code) {
  unsigned long long ga = (unsigned long long)gbase;
  v4u g0;
  g0[0] = 1u;                                   // count=1, user mode
  g0[1] = lds_addr;                             // LDS byte address
  g0[2] = (unsigned)(ga & 0xFFFFFFFFu);         // global_addr[31:0]
  g0[3] = (unsigned)((ga >> 32) & 0x01FFFFFFu) | (2u << 30);  // addr[56:32], type=2
  v8i g1;
  g1[0] = (int)((1u << 16) | (1u << 20) | (pad_int_code << 22) | (pad_amt_code << 25));
  g1[1] = (int)((row_elems & 0xFFFFu) << 16);   // tensor_dim0[15:0]
  g1[2] = (int)(((row_elems >> 16) & 0xFFFFu) | ((rows_valid & 0xFFFFu) << 16));
  g1[3] = (int)(((rows_valid >> 16) & 0xFFFFu) | ((row_elems & 0xFFFFu) << 16)); // tile_dim0 = row_elems
  g1[4] = (int)(tile_rows & 0xFFFFu);           // tile_dim1 (tile_dim2=0)
  g1[5] = (int)row_stride;                      // tensor_dim0_stride[31:0]
  g1[6] = 0;                                    // stride0[47:32]=0, stride1[15:0]=0
  g1[7] = 0;
  v4i gz4 = {0, 0, 0, 0};
  v8i gz8 = {0, 0, 0, 0, 0, 0, 0, 0};
  // clang-23 / amdgpu-toolchain 6-arg form: (g0, g1, g2, g3, g4, cpol)
  __builtin_amdgcn_tensor_load_to_lds(g0, g1, gz4, gz4, gz8, 0);
}

// ---------------- WMMA bf16 GEMM: out = act(A@Bt^T + bias) ------------------
// A: MxK f32 row-major. Bt: N x K bf16 (pre-transposed weights, N padded to 64,
// K multiple of 32 -- guaranteed by caller). Output f32 (C) or bf16 (C16).
__global__ __launch_bounds__(128) void gemm_bias_act(
    const float* __restrict__ A, const bf16* __restrict__ Bt,
    const float* __restrict__ bias, float* __restrict__ C, bf16* __restrict__ C16,
    int M, int N, int K, int act) {
  __shared__ bf16 As[64 * 40];   // 64 rows x 32 K, stride 40
  __shared__ bf16 Bs[64 * 40];   // 64 cols x 32 K, stride 40
  const int tid  = threadIdx.x;
  const int lane = tid & 31;
  const int wid  = tid >> 5;
  const int gm = blockIdx.y * 64;
  const int gn = blockIdx.x * 64;
  const int wr = (wid >> 1) * 32;
  const int wc = (wid & 1) * 32;
  const int sr = tid >> 2;        // 0..31 staging row
  const int sc = (tid & 3) * 8;   // 0,8,16,24 staging col (elements)

  v8f acc[2][2];
  v8f zf = {};
  for (int i = 0; i < 2; ++i) for (int j = 0; j < 2; ++j) acc[i][j] = zf;

  for (int k0 = 0; k0 < K; k0 += 32) {
    // stage A: two float4 loads -> 8 bf16 -> one b128 LDS store per thread
#pragma unroll
    for (int p = 0; p < 2; ++p) {
      int r = sr + p * 32;
      float4 f0 = {0.f, 0.f, 0.f, 0.f}, f1 = f0;
      if (gm + r < M) {
        const float4* s = (const float4*)(A + (size_t)(gm + r) * K + k0 + sc);
        f0 = s[0]; f1 = s[1];
      }
      Pack8 pk;
      pk.e[0] = f2bf(f0.x); pk.e[1] = f2bf(f0.y); pk.e[2] = f2bf(f0.z); pk.e[3] = f2bf(f0.w);
      pk.e[4] = f2bf(f1.x); pk.e[5] = f2bf(f1.y); pk.e[6] = f2bf(f1.z); pk.e[7] = f2bf(f1.w);
      *(v8bf*)(&As[r * 40 + sc]) = pk.v;
    }
    // stage B: raw 16B copies from pre-transposed bf16 weights (no guards needed)
#pragma unroll
    for (int p = 0; p < 2; ++p) {
      int n = sr + p * 32;
      *(v8bf*)(&Bs[n * 40 + sc]) = *(const v8bf*)(Bt + (size_t)(gn + n) * K + k0 + sc);
    }
    if (k0 + 32 < K)  // software prefetch of next A tile -> global_prefetch_b8
      __builtin_prefetch(A + (size_t)(gm + sr) * K + k0 + 32 + sc, 0, 0);
    __syncthreads();
    v16bf a0 = load_frag(As, 40, wr, 0);
    v16bf a1 = load_frag(As, 40, wr + 16, 0);
    v16bf b0 = load_frag(Bs, 40, wc, 0);
    v16bf b1 = load_frag(Bs, 40, wc + 16, 0);
    acc[0][0] = __builtin_amdgcn_wmma_f32_16x16x32_bf16(false, a0, false, b0, (short)0, acc[0][0], false, false);
    acc[0][1] = __builtin_amdgcn_wmma_f32_16x16x32_bf16(false, a0, false, b1, (short)0, acc[0][1], false, false);
    acc[1][0] = __builtin_amdgcn_wmma_f32_16x16x32_bf16(false, a1, false, b0, (short)0, acc[1][0], false, false);
    acc[1][1] = __builtin_amdgcn_wmma_f32_16x16x32_bf16(false, a1, false, b1, (short)0, acc[1][1], false, false);
    __syncthreads();
  }

  const int rofs = (lane >> 4) << 3;
  const int cofs = lane & 15;
  for (int ti = 0; ti < 2; ++ti)
    for (int tj = 0; tj < 2; ++tj)
      for (int e = 0; e < 8; ++e) {
        int m = gm + wr + ti * 16 + e + rofs;
        int n = gn + wc + tj * 16 + cofs;
        if (m < M && n < N) {
          float v = acc[ti][tj][e];
          if (bias) v += bias[n];
          if (act == 1) v = 0.5f * v * (1.0f + erff(v * 0.70710678118654752f));
          if (C16) C16[(size_t)m * N + n] = f2bf(v);
          else     C[(size_t)m * N + n] = v;
        }
      }
}

// ---------------- weight transpose + bf16 convert: out[n][k] = in[k][n] ------
__global__ __launch_bounds__(256) void transpose_f32_to_bf16(
    const float* __restrict__ in, bf16* __restrict__ out, int K, int N, int Npad) {
  __shared__ float t[16][17];
  int kb = blockIdx.y * 16, nb = blockIdx.x * 16;
  int tx = threadIdx.x & 15, ty = threadIdx.x >> 4;
  int k = kb + ty, n = nb + tx;
  t[ty][tx] = (k < K && n < N) ? in[(size_t)k * N + n] : 0.f;  // pad cols -> 0
  __syncthreads();
  int ko = kb + tx, no = nb + ty;
  if (no < Npad && ko < K) out[(size_t)no * K + ko] = f2bf(t[tx][ty]);
}

// ---------------- Fused attention per (b,r,h) -------------------------------
// qv16: (B*N, 2*DIM) bf16; k16: (R*N, DIM) bf16. obuf: (B, R, H*N*HD) f32.
__global__ __launch_bounds__(256) void attn_fused(
    const bf16* __restrict__ qv16, const bf16* __restrict__ k16,
    float* __restrict__ obuf) {
  extern __shared__ char smem_raw[];
  bf16* qs   = (bf16*)smem_raw;        // NP x QSS
  bf16* ks   = qs  + NP * QSS;         // NP x QSS
  bf16* vtr  = ks  + NP * QSS;         // HD x VST  (v transposed)
  bf16* Ab   = vtr + HD * VST;         // NP x AST
  bf16* Sall = Ab  + NP * AST;         // 8 waves x 16 x AST

  const int h = blockIdx.x, r = blockIdx.y, b = blockIdx.z;
  const int tid  = threadIdx.x;
  const int lane = tid & 31;
  const int wid  = tid >> 5;

  // Tensor Data Mover: stage q and k tiles (rows>=197 auto-zeroed by OOB reads,
  // LDS pad 16B after each 128B row -> stride 144B == QSS elements).
  if (wid == 0) {
    tdm_load_2d_bf16((unsigned)(unsigned long long)qs,
                     qv16 + (size_t)(b * NTOK) * (2 * DIM) + h * HD,
                     NTOK, HD, 2 * DIM, NP, /*pad 32 dw*/4, /*4 dw*/3);
    tdm_load_2d_bf16((unsigned)(unsigned long long)ks,
                     k16 + (size_t)(r * NTOK) * DIM + h * HD,
                     NTOK, HD, DIM, NP, 4, 3);
    __builtin_amdgcn_s_wait_tensorcnt(0);
  }
  // stage v transposed: one 16B load + 8 u16 LDS stores per chunk
  for (int idx = tid; idx < NP * (HD / 8); idx += 256) {
    int k = idx >> 3;
    int d0 = (idx & 7) * 8;
    Pack8 val;
    v8bf z = {};
    val.v = z;
    if (k < NTOK)
      val.v = *(const v8bf*)(qv16 + (size_t)(b * NTOK + k) * (2 * DIM) + DIM + h * HD + d0);
#pragma unroll
    for (int i = 0; i < 8; ++i) vtr[(d0 + i) * VST + k] = val.e[i];
  }
  __syncthreads();

  // Phase 1: A = 0.125 * q k^T  (14x14 tiles, K=64 in 2 WMMA)
  for (int t = wid; t < 14 * 14; t += 8) {
    int ti = t / 14, tj = t % 14;
    v8f acc = {};
    for (int kd = 0; kd < 2; ++kd) {
      v16bf a  = load_frag(qs, QSS, ti * 16, kd * 32);
      v16bf bb = load_frag(ks, QSS, tj * 16, kd * 32);
      acc = __builtin_amdgcn_wmma_f32_16x16x32_bf16(false, a, false, bb, (short)0, acc, false, false);
    }
    int row = ti * 16 + ((lane >> 4) << 3);
    int col = tj * 16 + (lane & 15);
#pragma unroll
    for (int e = 0; e < 8; ++e)
      Ab[(row + e) * AST + col] = f2bf(acc[e] * 0.125f);
  }
  __syncthreads();

  // Phase 2+3 per wave: S_i = A_i A^T (16x224), then O_i = S_i v (16x64)
  bf16* Sw = Sall + wid * 16 * AST;
  const int rofs = (lane >> 4) << 3;
  const int cofs = lane & 15;
  for (int i = wid; i < 14; i += 8) {
    v8f sacc[14];
    v8f zf = {};
#pragma unroll
    for (int j = 0; j < 14; ++j) sacc[j] = zf;
    for (int kst = 0; kst < 7; ++kst) {
      v16bf a = load_frag(Ab, AST, i * 16, kst * 32);
#pragma unroll
      for (int j = 0; j < 14; ++j) {
        v16bf bb = load_frag(Ab, AST, j * 16, kst * 32);
        sacc[j] = __builtin_amdgcn_wmma_f32_16x16x32_bf16(false, a, false, bb, (short)0, sacc[j], false, false);
      }
    }
#pragma unroll
    for (int j = 0; j < 14; ++j)
#pragma unroll
      for (int e = 0; e < 8; ++e)
        Sw[(rofs + e) * AST + j * 16 + cofs] = f2bf(sacc[j][e]);

    for (int dt = 0; dt < 4; ++dt) {
      v8f oacc = {};
      for (int kst = 0; kst < 7; ++kst) {
        v16bf a  = load_frag(Sw, AST, 0, kst * 32);
        v16bf bb = load_frag(vtr, VST, dt * 16, kst * 32);
        oacc = __builtin_amdgcn_wmma_f32_16x16x32_bf16(false, a, false, bb, (short)0, oacc, false, false);
      }
#pragma unroll
      for (int e = 0; e < 8; ++e) {
        int n = i * 16 + rofs + e;
        if (n < NTOK) {
          int d = dt * 16 + cofs;
          size_t off = (((size_t)(b * NB_REFC + r)) * HEADS + h) * (NTOK * HD) + (size_t)n * HD + d;
          obuf[off] = oacc[e];
        }
      }
    }
  }
}

// ---------------- small scalar kernels ----------------
__global__ void mask_k(const float* __restrict__ e, float* __restrict__ mask) {
  int idx = blockIdx.x * blockDim.x + threadIdx.x;
  if (idx >= BATCH * NTOK) return;
  int b = idx / NTOK, n = idx % NTOK;
  const float* ex = e + (size_t)(b * NTOK + n) * 96;
  float nx = 0.f;
  for (int c = 0; c < 96; ++c) nx += ex[c] * ex[c];
  nx = fmaxf(sqrtf(nx), 1e-8f);
  float s = 0.f;
  for (int r = 0; r < NB_REFC; ++r) {
    const float* er = e + (size_t)((BATCH + r) * NTOK + n) * 96;
    float d = 0.f, nr = 0.f;
    for (int c = 0; c < 96; ++c) { d += ex[c] * er[c]; nr += er[c] * er[c]; }
    nr = fmaxf(sqrtf(nr), 1e-8f);
    s += d / (nx * nr);
  }
  float cc = (s / (float)NB_REFC + 1.0f) * 0.5f;
  mask[idx] = 0.5f + 0.5f * tanhf(25.0f * (cc - 0.5f));
}

__global__ void mask_apply_k(const float* __restrict__ x, const float* __restrict__ mask,
                             float* __restrict__ out) {
  int idx = blockIdx.x * blockDim.x + threadIdx.x;
  if (idx >= BATCH * NTOK * DIM) return;
  out[idx] = x[idx] * mask[idx / DIM];
}

__global__ void mean_over_r(const float* __restrict__ obuf, float* __restrict__ t) {
  int idx = blockIdx.x * blockDim.x + threadIdx.x;
  const int F = NTOK * DIM;  // == HEADS*NTOK*HD: implements the raw reshape
  if (idx >= BATCH * F) return;
  int b = idx / F, f = idx % F;
  float s = 0.f;
  for (int r = 0; r < NB_REFC; ++r)
    s += obuf[(size_t)(b * NB_REFC + r) * F + f];
  t[idx] = s * (1.0f / (float)NB_REFC);
}

__global__ __launch_bounds__(256) void layernorm_k(
    const float* __restrict__ y, const float* __restrict__ g,
    const float* __restrict__ bta, float* __restrict__ out) {
  __shared__ float red[256];
  int row = blockIdx.x;
  const float* yr = y + (size_t)row * DIM;
  float s = 0.f;
  for (int c = threadIdx.x; c < DIM; c += 256) s += yr[c];
  red[threadIdx.x] = s; __syncthreads();
  for (int st = 128; st > 0; st >>= 1) {
    if (threadIdx.x < st) red[threadIdx.x] += red[threadIdx.x + st];
    __syncthreads();
  }
  float mu = red[0] * (1.0f / (float)DIM);
  __syncthreads();
  float v = 0.f;
  for (int c = threadIdx.x; c < DIM; c += 256) { float d = yr[c] - mu; v += d * d; }
  red[threadIdx.x] = v; __syncthreads();
  for (int st = 128; st > 0; st >>= 1) {
    if (threadIdx.x < st) red[threadIdx.x] += red[threadIdx.x + st];
    __syncthreads();
  }
  float inv = rsqrtf(red[0] * (1.0f / (float)DIM) + 1e-5f);
  for (int c = threadIdx.x; c < DIM; c += 256)
    out[(size_t)row * DIM + c] = (yr[c] - mu) * inv * g[c] + bta[c];
}

__global__ void final_k(const float* __restrict__ x, const float* __restrict__ corr,
                        const float* __restrict__ mask, float* __restrict__ out) {
  int idx = blockIdx.x * blockDim.x + threadIdx.x;
  const int NM  = BATCH * NTOK * DIM;
  const int TOT = (BATCH + NB_REFC) * NTOK * DIM;
  if (idx >= TOT) return;
  if (idx < NM) {
    float m = mask[idx / DIM];
    out[idx] = m * corr[idx] + (1.0f - m) * x[idx];
  } else {
    out[idx] = x[idx];  // ref passthrough
  }
}

// ---------------- orchestration ----------------
extern "C" void kernel_launch(void* const* d_in, const int* in_sizes, int n_in,
                              void* d_out, int out_size, void* d_ws, size_t ws_size,
                              hipStream_t stream) {
  const float* x      = (const float*)d_in[0];
  const float* qv_w   = (const float*)d_in[1];
  const float* k_w    = (const float*)d_in[2];
  const float* proj_w = (const float*)d_in[3];
  const float* proj_b = (const float*)d_in[4];
  const float* ln_g   = (const float*)d_in[5];
  const float* ln_b   = (const float*)d_in[6];
  const float* fc1_w  = (const float*)d_in[7];
  const float* fc1_b  = (const float*)d_in[8];
  const float* fc2_w  = (const float*)d_in[9];
  const float* fc2_b  = (const float*)d_in[10];
  float* out = (float*)d_out;

  const int ROWS_ALL = (BATCH + NB_REFC) * NTOK;  // 3546
  const int ROWS_B   = BATCH * NTOK;              // 1576
  const int ROWS_R   = NB_REFC * NTOK;            // 1970

  float* ws = (float*)d_ws;
  float* h1    = ws; ws += (size_t)ROWS_ALL * 192;
  float* ebuf  = ws; ws += (size_t)ROWS_ALL * 96;
  float* maskb = ws; ws += (size_t)ROWS_B;
  float* xcur  = ws; ws += (size_t)ROWS_B * DIM;
  float* obuf  = ws; ws += (size_t)BATCH * NB_REFC * HEADS * NTOK * HD;
  float* tbuf  = ws; ws += (size_t)ROWS_B * DIM;
  float* ybuf  = ws; ws += (size_t)ROWS_B * DIM;
  ws += (8 - ((size_t)(ws - (float*)d_ws) & 7)) & 7;  // 32B-align bf16 region
  bf16* bw = (bf16*)ws;
  bf16* qv_wt   = bw; bw += (size_t)1536 * DIM;       // (2D)xD transposed
  bf16* k_wt    = bw; bw += (size_t)DIM * DIM;
  bf16* proj_wt = bw; bw += (size_t)DIM * DIM;
  bf16* fc1_wt  = bw; bw += (size_t)192 * DIM;
  bf16* fc2_wt  = bw; bw += (size_t)128 * 192;        // 96 padded to 128 rows
  bf16* qvb16   = bw; bw += (size_t)ROWS_B * 2 * DIM;
  bf16* k16     = bw; bw += (size_t)ROWS_R * DIM;
  (void)ws_size; (void)in_sizes; (void)n_in; (void)out_size;

  auto ggrid = [](int M, int N) { return dim3((unsigned)((N + 63) / 64), (unsigned)((M + 63) / 64)); };
  auto tgrid = [](int K, int Npad) { return dim3((unsigned)((Npad + 15) / 16), (unsigned)((K + 15) / 16)); };
  dim3 gblk(128);

  // pre-transpose + convert all weights to bf16 (N x K layout)
  transpose_f32_to_bf16<<<tgrid(DIM, 1536), 256, 0, stream>>>(qv_w,   qv_wt,   DIM, 1536, 1536);
  transpose_f32_to_bf16<<<tgrid(DIM, DIM),  256, 0, stream>>>(k_w,    k_wt,    DIM, DIM,  DIM);
  transpose_f32_to_bf16<<<tgrid(DIM, DIM),  256, 0, stream>>>(proj_w, proj_wt, DIM, DIM,  DIM);
  transpose_f32_to_bf16<<<tgrid(DIM, 192),  256, 0, stream>>>(fc1_w,  fc1_wt,  DIM, 192,  192);
  transpose_f32_to_bf16<<<tgrid(192, 128),  256, 0, stream>>>(fc2_w,  fc2_wt,  192, 96,   128);

  // projector mask path
  gemm_bias_act<<<ggrid(ROWS_ALL, 192), gblk, 0, stream>>>(x, fc1_wt, fc1_b, h1, nullptr, ROWS_ALL, 192, DIM, 1);
  gemm_bias_act<<<ggrid(ROWS_ALL, 96),  gblk, 0, stream>>>(h1, fc2_wt, fc2_b, ebuf, nullptr, ROWS_ALL, 96, 192, 0);
  mask_k<<<(ROWS_B + 255) / 256, 256, 0, stream>>>(ebuf, maskb);

  // k = ref @ k_w (bf16 output, consumed by TDM staging)
  const float* refp = x + (size_t)BATCH * NTOK * DIM;
  gemm_bias_act<<<ggrid(ROWS_R, DIM), gblk, 0, stream>>>(refp, k_wt, nullptr, nullptr, k16, ROWS_R, DIM, DIM, 0);

  // x0 = xm * mask
  mask_apply_k<<<(ROWS_B * DIM + 255) / 256, 256, 0, stream>>>(x, maskb, xcur);

  const size_t attn_lds = (size_t)(NP * QSS * 2 + HD * VST + NP * AST + 8 * 16 * AST) * 2u;
  for (int s = 0; s < STEPSC; ++s) {
    gemm_bias_act<<<ggrid(ROWS_B, 2 * DIM), gblk, 0, stream>>>(xcur, qv_wt, nullptr, nullptr, qvb16, ROWS_B, 2 * DIM, DIM, 0);
    attn_fused<<<dim3(HEADS, NB_REFC, BATCH), 256, attn_lds, stream>>>(qvb16, k16, obuf);
    mean_over_r<<<(ROWS_B * DIM + 255) / 256, 256, 0, stream>>>(obuf, tbuf);
    gemm_bias_act<<<ggrid(ROWS_B, DIM), gblk, 0, stream>>>(tbuf, proj_wt, proj_b, ybuf, nullptr, ROWS_B, DIM, DIM, 0);
    layernorm_k<<<ROWS_B, 256, 0, stream>>>(ybuf, ln_g, ln_b, xcur);
  }

  final_k<<<((BATCH + NB_REFC) * NTOK * DIM + 255) / 256, 256, 0, stream>>>(x, xcur, maskb, out);
}
